// PathEmbedding_4561255268491
// MI455X (gfx1250) — matrix operand order repaired and checked by reading, not compiled, not run on hardware
//
#include <hip/hip_runtime.h>
#include <hip/hip_bf16.h>

typedef __attribute__((ext_vector_type(16))) _Float16 v16h;
typedef __attribute__((ext_vector_type(8)))  float    v8f;

union F16Frag { v16h v; unsigned u[8]; };

// ---- fast device math ------------------------------------------------------

__device__ inline float fast_sigmoid(float x) {
  return __builtin_amdgcn_rcpf(1.f + __expf(-x));
}
__device__ inline float fast_tanh(float x) {
  // tanh(x) = 1 - 2/(e^{2x}+1); exact at +/-inf limits
  return 1.f - 2.f * __builtin_amdgcn_rcpf(__expf(2.f * x) + 1.f);
}

// ---- WMMA helpers (gfx1250, wave32, 16x16x32 f16 -> f32) -------------------

__device__ inline v8f wmma_f16(v16h a, v16h b, v8f c) {
  return __builtin_amdgcn_wmma_f32_16x16x32_f16(
      /*neg_a=*/false, a, /*neg_b=*/false, b,
      /*c_mod=*/(short)0, c, /*reuse_a=*/false, /*reuse_b=*/false);
}

// A-matrix 16x32 f16 fragment (row-major source, leading dim `ld` halves).
// lane<16 holds K {0..7,16..23}; lane>=16 holds K {8..15,24..31}; row = row0 + lane%16.
__device__ inline v16h load_afrag(const _Float16* p, int ld, int row0, int k0, int lane) {
  int r  = row0 + (lane & 15);
  int kb = (lane >> 4) * 8;
  const _Float16* base = p + r * ld + k0 + kb;
  F16Frag f;
#pragma unroll
  for (int j = 0; j < 8; ++j) {
    int K = (j & 3) * 2 + (j >> 2) * 16;           // halves (K, K+1)
    f.u[j] = *(const unsigned*)(base + K);
  }
  return f.v;
}

// B-matrix 32x16 f16 fragment. B[k][n] = W[n][k] (W row-major NxK).
// lane holds column n = n0 + lane%16; lane<16 -> K {0..7,16..23}, lane>=16 -> K {8..15,24..31}.
__device__ inline v16h load_bfrag(const _Float16* p, int ld, int n0, int k0, int lane) {
  int n  = n0 + (lane & 15);
  int kb = (lane >> 4) * 8;
  const _Float16* base = p + n * ld + k0 + kb;
  F16Frag f;
#pragma unroll
  for (int j = 0; j < 8; ++j)
    f.u[j] = *(const unsigned*)(base + (j & 3) * 2 + (j >> 2) * 16);
  return f.v;
}

// C/D 16x16 f32: VGPR v, lanes 0-15 -> (M=v, N=lane); lanes 16-31 -> (M=v+8, N=lane-16)
__device__ inline void store_acc(float* out, int ldo, int m0, int n0, v8f acc, int lane) {
  int n  = n0 + (lane & 15);
  int mb = m0 + (lane >> 4) * 8;
#pragma unroll
  for (int v = 0; v < 8; ++v) out[(mb + v) * ldo + n] = acc[v];
}

// ---- Kernel 1: f32 -> f16 weight conversion (192 KB, L2-resident) ----------

__global__ void convert_weights(const float* Wi, const float* Wh,
                                const float* Ww, const float* Wu, _Float16* out) {
  int i = blockIdx.x * blockDim.x + threadIdx.x;
  if (i < 32768)       out[i] = (_Float16)Wi[i];
  else if (i < 65536)  out[i] = (_Float16)Wh[i - 32768];
  else if (i < 81920)  out[i] = (_Float16)Ww[i - 65536];
  else if (i < 98304)  out[i] = (_Float16)Wu[i - 81920];
}

// ---- Kernel 2: fused embedding-gather + GRU scan (WMMA) --------------------

#define BT   32     // sequences per block
#define HD   128    // H = D
#define GD   256    // 2H
#define LSEQ 32

__global__ __launch_bounds__(256) void gru_kernel(
    const int* __restrict__ paths, const int* __restrict__ paths_mask,
    const float* __restrict__ emb,
    const float* __restrict__ bi, const float* __restrict__ bh,
    const float* __restrict__ bw, const float* __restrict__ bu,
    const _Float16* __restrict__ wf16, float* __restrict__ out_ws) {

  __shared__ __align__(16) _Float16 hf16[BT * HD];   // 8 KB
  __shared__ __align__(16) _Float16 xf16[BT * HD];   // 8 KB
  __shared__ __align__(16) float bufA[BT * GD];      // 32 KB (h2h raw -> gates z|r)
  __shared__ __align__(16) float bufB[BT * GD];      // 32 KB (xi raw -> hu|xw raw)
  __shared__ float s_bi[GD], s_bh[GD], s_bw[HD], s_bu[HD];
  __shared__ int   s_len[BT];

  const int tid  = threadIdx.x;
  const int lane = tid & 31;
  const int wave = tid >> 5;
  const int base = blockIdx.x * BT;

  const _Float16* Wi16 = wf16;
  const _Float16* Wh16 = wf16 + 32768;
  const _Float16* Ww16 = wf16 + 65536;
  const _Float16* Wu16 = wf16 + 81920;

  // thread owns h row = tid>>3, cols (tid&7)*16 .. +15
  const int urow = tid >> 3;
  const int useg = tid & 7;
  float hreg[16];
#pragma unroll
  for (int c = 0; c < 16; ++c) hreg[c] = 0.f;

  if (tid < GD) { s_bi[tid] = bi[tid]; s_bh[tid] = bh[tid]; }
  if (tid < HD) { s_bw[tid] = bw[tid]; s_bu[tid] = bu[tid]; }
  if (tid < BT) s_len[tid] = paths_mask[base + tid];
  for (int i = tid; i < BT * HD; i += 256) hf16[i] = (_Float16)0.f;
  // gather x_0
  {
    int tok = paths[(base + urow) * LSEQ + 0];
    const float* src = emb + tok * HD + useg * 16;
    _Float16* dst = xf16 + urow * HD + useg * 16;
#pragma unroll
    for (int c = 0; c < 16; ++c) dst[c] = (_Float16)src[c];
  }
  __syncthreads();

  for (int t = 0; t < LSEQ; ++t) {
    // -- WMMA phase 1: bufA = h@Wh^T, bufB = x@Wi^T  (32x256 each) --
#pragma unroll
    for (int nt2 = 0; nt2 < 2; ++nt2) {
      int nt = wave * 2 + nt2;                       // 0..15
      const _Float16* wh = Wh16 + nt * 16 * HD;
      const _Float16* wi = Wi16 + nt * 16 * HD;
      v8f aH0 = {}, aH1 = {}, aX0 = {}, aX1 = {};
#pragma unroll
      for (int k = 0; k < 4; ++k) {
        int k0 = k * 32;
        v16h bH = load_bfrag(wh, HD, 0, k0, lane);
        v16h bI = load_bfrag(wi, HD, 0, k0, lane);
        v16h a0 = load_afrag(hf16, HD, 0,  k0, lane);
        v16h a1 = load_afrag(hf16, HD, 16, k0, lane);
        v16h x0 = load_afrag(xf16, HD, 0,  k0, lane);
        v16h x1 = load_afrag(xf16, HD, 16, k0, lane);
        aH0 = wmma_f16(a0, bH, aH0);  aH1 = wmma_f16(a1, bH, aH1);
        aX0 = wmma_f16(x0, bI, aX0);  aX1 = wmma_f16(x1, bI, aX1);
      }
      store_acc(bufA, GD, 0,  nt * 16, aH0, lane);
      store_acc(bufA, GD, 16, nt * 16, aH1, lane);
      store_acc(bufB, GD, 0,  nt * 16, aX0, lane);
      store_acc(bufB, GD, 16, nt * 16, aX1, lane);
    }
    __syncthreads();

    // -- fused LN(h2h+bh), LN(xi+bi), gates = sigmoid(.+.) -> bufA = [z|r] --
    {
      const int row = tid >> 3, seg = tid & 7;       // 32-col slice
      float* arow = bufA + row * GD;
      float* brow = bufB + row * GD;
      float sa = 0.f, qa = 0.f, sb = 0.f, qb = 0.f;
      for (int c = 0; c < 32; ++c) {
        int cc = seg * 32 + c;
        float a = arow[cc] + s_bh[cc];
        float b = brow[cc] + s_bi[cc];
        sa += a; qa += a * a; sb += b; qb += b * b;
      }
#pragma unroll
      for (int m = 1; m < 8; m <<= 1) {
        sa += __shfl_xor(sa, m, 32);  qa += __shfl_xor(qa, m, 32);
        sb += __shfl_xor(sb, m, 32);  qb += __shfl_xor(qb, m, 32);
      }
      float ma = sa * (1.f / 256.f), mb = sb * (1.f / 256.f);
      float ra = rsqrtf(qa * (1.f / 256.f) - ma * ma + 1e-5f);
      float rb = rsqrtf(qb * (1.f / 256.f) - mb * mb + 1e-5f);
      for (int c = 0; c < 32; ++c) {
        int cc = seg * 32 + c;
        float a = (arow[cc] + s_bh[cc] - ma) * ra;
        float b = (brow[cc] + s_bi[cc] - mb) * rb;
        arow[cc] = fast_sigmoid(a + b);
      }
    }
    __syncthreads();

    // -- WMMA phase 2: bufB[:, :128] = h@Wu^T, bufB[:, 128:] = x@Ww^T --
#pragma unroll
    for (int nt2 = 0; nt2 < 2; ++nt2) {
      int nt = wave * 2 + nt2;                       // 0..15 logical
      const _Float16* W;  const _Float16* A;  int colb;
      if (nt < 8) { W = Wu16 + nt * 16 * HD;        A = hf16; colb = nt * 16; }
      else        { W = Ww16 + (nt - 8) * 16 * HD;  A = xf16; colb = 128 + (nt - 8) * 16; }
      v8f c0 = {}, c1 = {};
#pragma unroll
      for (int k = 0; k < 4; ++k) {
        int k0 = k * 32;
        v16h b  = load_bfrag(W, HD, 0, k0, lane);
        v16h a0 = load_afrag(A, HD, 0,  k0, lane);
        v16h a1 = load_afrag(A, HD, 16, k0, lane);
        c0 = wmma_f16(a0, b, c0);  c1 = wmma_f16(a1, b, c1);
      }
      store_acc(bufB, GD, 0,  colb, c0, lane);
      store_acc(bufB, GD, 16, colb, c1, lane);
    }
    __syncthreads();

    // -- fused LN(hu+bu), LN(xw+bw), h-update, emit, next-token gather --
    {
      float* mrow = bufB + urow * GD;
      const float* grow = bufA + urow * GD;
      float su = 0.f, qu = 0.f, sw = 0.f, qw = 0.f;
      float huv[16], xwv[16];
#pragma unroll
      for (int c = 0; c < 16; ++c) {
        int cc = useg * 16 + c;
        float a = mrow[cc]       + s_bu[cc];
        float b = mrow[128 + cc] + s_bw[cc];
        huv[c] = a; xwv[c] = b;
        su += a; qu += a * a; sw += b; qw += b * b;
      }
#pragma unroll
      for (int m = 1; m < 8; m <<= 1) {
        su += __shfl_xor(su, m, 32);  qu += __shfl_xor(qu, m, 32);
        sw += __shfl_xor(sw, m, 32);  qw += __shfl_xor(qw, m, 32);
      }
      float mu = su * (1.f / 128.f), mw = sw * (1.f / 128.f);
      float ru = rsqrtf(qu * (1.f / 128.f) - mu * mu + 1e-5f);
      float rw = rsqrtf(qw * (1.f / 128.f) - mw * mw + 1e-5f);
      int  len  = s_len[urow];
      bool emit = (t == len - 1);
#pragma unroll
      for (int c = 0; c < 16; ++c) {
        int cc = useg * 16 + c;
        float z  = grow[cc];
        float r  = grow[128 + cc];
        float hu = (huv[c] - mu) * ru;
        float xw = (xwv[c] - mw) * rw;
        float hh = fast_tanh(xw + r * hu);
        float hn = (1.f - z) * hreg[c] + z * hh;
        hreg[c] = hn;
        hf16[urow * HD + cc] = (_Float16)hn;
        if (emit) out_ws[(base + urow) * HD + cc] = hn;
      }
      // prefetch next token's embedding row into xf16 (phase-2 done with it)
      if (t + 1 < LSEQ) {
        int tok = paths[(base + urow) * LSEQ + t + 1];
        const float* src = emb + tok * HD + useg * 16;
        _Float16* dst = xf16 + urow * HD + useg * 16;
#pragma unroll
        for (int c = 0; c < 16; ++c) dst[c] = (_Float16)src[c];
      }
    }
    __syncthreads();
  }
}

// ---- Kernel 3: fwd/bwd interleave + final LN * gamma + beta ----------------

__global__ __launch_bounds__(256) void finalize_kernel(
    const float* __restrict__ out_ws, const float* __restrict__ gamma,
    const float* __restrict__ beta, float* __restrict__ d_out) {
  __shared__ float red_s[8], red_q[8];
  int s = blockIdx.x;            // 0..8191 = bs*P row
  int tid = threadIdx.x;
  // res[:, p, :128] = out[p]; res[:, p, 128:] = out[p^1]
  float val = (tid < 128) ? out_ws[s * 128 + tid]
                          : out_ws[(s ^ 1) * 128 + (tid - 128)];
  float sum = val, sq = val * val;
#pragma unroll
  for (int m = 1; m < 32; m <<= 1) {
    sum += __shfl_xor(sum, m, 32);
    sq  += __shfl_xor(sq,  m, 32);
  }
  if ((tid & 31) == 0) { red_s[tid >> 5] = sum; red_q[tid >> 5] = sq; }
  __syncthreads();
  float ts = 0.f, tq = 0.f;
#pragma unroll
  for (int i = 0; i < 8; ++i) { ts += red_s[i]; tq += red_q[i]; }
  float mean = ts * (1.f / 256.f);
  float var  = tq * (1.f / 256.f) - mean * mean;
  float rs   = rsqrtf(var + 1e-5f);
  d_out[s * 256 + tid] = (val - mean) * rs * gamma[tid] + beta[tid];
}

// ---- Host launcher ---------------------------------------------------------

extern "C" void kernel_launch(void* const* d_in, const int* in_sizes, int n_in,
                              void* d_out, int out_size, void* d_ws, size_t ws_size,
                              hipStream_t stream) {
  const int*   paths = (const int*)d_in[0];    // (32,256,32)
  const int*   mask  = (const int*)d_in[1];    // (32,256)
  const float* emb   = (const float*)d_in[2];  // (301,128)
  const float* Wi    = (const float*)d_in[3];  // (256,128)
  const float* bi    = (const float*)d_in[4];
  const float* Wh    = (const float*)d_in[5];  // (256,128)
  const float* bh    = (const float*)d_in[6];
  const float* Ww    = (const float*)d_in[7];  // (128,128)
  const float* bw    = (const float*)d_in[8];
  const float* Wu    = (const float*)d_in[9];  // (128,128)
  const float* bu    = (const float*)d_in[10];
  const float* gamma = (const float*)d_in[11];
  const float* beta  = (const float*)d_in[12];

  _Float16* wf16   = (_Float16*)d_ws;                       // 196608 B
  float*    out_ws = (float*)((char*)d_ws + 196608);        // 8192*128*4 B

  convert_weights<<<384, 256, 0, stream>>>(Wi, Wh, Ww, Wu, wf16);
  gru_kernel<<<256, 256, 0, stream>>>(paths, mask, emb, bi, bh, bw, bu, wf16, out_ws);
  finalize_kernel<<<8192, 256, 0, stream>>>(out_ws, gamma, beta, (float*)d_out);
}
